// ScatterNet_21706764714521
// MI455X (gfx1250) — compile-verified, matrix operand order checked by reading, not run on hardware
//
#include <hip/hip_runtime.h>
#include <hip/hip_bf16.h>
#include <math.h>

typedef _Float16 f16;
typedef __attribute__((ext_vector_type(8)))  _Float16 v8h;
typedef __attribute__((ext_vector_type(16))) _Float16 v16h;
typedef __attribute__((ext_vector_type(8)))  float    v8f;

// ---------------------------------------------------------------------------
// Generic tiled WMMA GEMM:  C = epi(A[MxK] x B + bias + res) (optionally *mask)
//   opB=0: B is KxN row-major.  opB=1: B is NxK row-major (computes A @ B^T).
//   epi: 0=none 1=elu 2=relu 3=sigmoid(v*oscale)*mask
//   out16: 1 -> f16 output, 0 -> f32 output
// Block = 128 threads = 4 waves (2x2); each wave owns a 32x32 sub-tile
// (2x2 fragments of v_wmma_f32_16x16x32_f16).
// Both LDS tiles are K-contiguous (As[m][k], Bs[n][k]; rows padded to 72
// halves = 144B so fragment reads are aligned conflict-free ds_load_b128),
// double-buffered in LDS with register-staged global prefetch:
// one barrier per k-tile, no register pipeline copies.
// ---------------------------------------------------------------------------

__device__ __forceinline__ void g_load_tiles(
    const f16* __restrict__ A, const f16* __restrict__ Bm,
    int M, int N, int K, int lda, int ldb,
    int m0, int n0, int k0, int opB, int tid,
    v8h ra[2], v8h rb[2])
{
    #pragma unroll
    for (int t = 0; t < 2; ++t) {
        const int c  = tid + t * 128;
        const int r  = c >> 2;
        const int co = (c & 3) << 3;
        const int gm = m0 + r, gk = k0 + co;
        v8h v;
        if (gm < M && gk + 8 <= K) {
            v = *(const v8h*)(A + (long long)gm * lda + gk);
        } else {
            #pragma unroll
            for (int u = 0; u < 8; ++u)
                v[u] = (gm < M && gk + u < K) ? A[(long long)gm * lda + gk + u] : (f16)0.f;
        }
        ra[t] = v;
    }
    #pragma unroll
    for (int t = 0; t < 2; ++t) {
        const int c = tid + t * 128;
        v8h v;
        if (opB == 0) {
            // B KxN row-major: read 8 contiguous along n for row gk
            const int r  = c >> 3;            // k within tile
            const int co = (c & 7) << 3;      // n within tile
            const int gk = k0 + r, gn = n0 + co;
            if (gk < K && gn + 8 <= N) {
                v = *(const v8h*)(Bm + (long long)gk * ldb + gn);
            } else {
                #pragma unroll
                for (int u = 0; u < 8; ++u)
                    v[u] = (gk < K && gn + u < N) ? Bm[(long long)gk * ldb + gn + u] : (f16)0.f;
            }
        } else {
            // B NxK row-major: read 8 contiguous along k for row gn
            const int n  = c >> 2;            // n within tile
            const int ko = (c & 3) << 3;      // k within tile
            const int gn = n0 + n, gk = k0 + ko;
            if (gn < N && gk + 8 <= K) {
                v = *(const v8h*)(Bm + (long long)gn * ldb + gk);
            } else {
                #pragma unroll
                for (int u = 0; u < 8; ++u)
                    v[u] = (gn < N && gk + u < K) ? Bm[(long long)gn * ldb + gk + u] : (f16)0.f;
            }
        }
        rb[t] = v;
    }
}

__device__ __forceinline__ void stage_tiles(
    f16 (*As)[72], f16 (*Bs)[72], int tid, int opB,
    const v8h ra[2], const v8h rb[2])
{
    #pragma unroll
    for (int t = 0; t < 2; ++t) {
        const int c  = tid + t * 128;
        const int r  = c >> 2;
        const int co = (c & 3) << 3;
        *(v8h*)&As[r][co] = ra[t];
    }
    if (opB == 0) {
        #pragma unroll
        for (int t = 0; t < 2; ++t) {       // transpose into Bs[n][k]
            const int c  = tid + t * 128;
            const int r  = c >> 3;           // k
            const int co = (c & 7) << 3;     // n
            #pragma unroll
            for (int u = 0; u < 8; ++u) Bs[co + u][r] = rb[t][u];
        }
    } else {
        #pragma unroll
        for (int t = 0; t < 2; ++t) {
            const int c  = tid + t * 128;
            const int n  = c >> 2;
            const int ko = (c & 3) << 3;
            *(v8h*)&Bs[n][ko] = rb[t];
        }
    }
}

__global__ __launch_bounds__(128)
void k_wmma_gemm(const f16* __restrict__ A, const f16* __restrict__ Bm,
                 const float* __restrict__ bias, const float* __restrict__ res,
                 void* __restrict__ Cp, const unsigned char* __restrict__ mask,
                 int M, int N, int K,
                 int lda, int ldb, int ldc, int ldres, int ldmask,
                 int opB, int epi, int out16, float oscale,
                 long long sA, long long sB, long long sC, long long sRes, long long sMask)
{
    const int bz = blockIdx.z;
    A  += (long long)bz * sA;
    Bm += (long long)bz * sB;
    if (res)  res  += (long long)bz * sRes;
    if (mask) mask += (long long)bz * sMask;

    const int m0   = blockIdx.y * 64;
    const int n0   = blockIdx.x * 64;
    const int tid  = threadIdx.x;
    const int lane = tid & 31;
    const int wave = tid >> 5;
    const int wm   = (wave >> 1) * 32;
    const int wn   = (wave & 1) * 32;

    __shared__ f16 As[2][64][72];   // [buf][m][k], 144B rows
    __shared__ f16 Bs[2][64][72];   // [buf][n][k]

    v8f acc[2][2] = {};

    const int mrow = lane & 15;
    const int kb   = (lane >> 4) << 3;   // 0 or 8
    const int nk   = (K + 31) >> 5;

    v8h ra[2], rb[2];
    // prologue: stage tile 0 into buffer 0, prefetch tile 1 into registers
    g_load_tiles(A, Bm, M, N, K, lda, ldb, m0, n0, 0, opB, tid, ra, rb);
    stage_tiles(As[0], Bs[0], tid, opB, ra, rb);
    if (nk > 1)
        g_load_tiles(A, Bm, M, N, K, lda, ldb, m0, n0, 32, opB, tid, ra, rb);
    __syncthreads();

    for (int kt = 0; kt < nk; ++kt) {
        const int buf = kt & 1;

        // ---- fragments: two aligned b128 LDS loads each ----
        v16h af[2], bf[2];
        #pragma unroll
        for (int f = 0; f < 2; ++f) {
            const int mr = wm + f * 16 + mrow;
            v8h a0 = *(const v8h*)&As[buf][mr][kb];
            v8h a1 = *(const v8h*)&As[buf][mr][kb + 16];
            af[f] = __builtin_shufflevector(a0, a1, 0,1,2,3,4,5,6,7,8,9,10,11,12,13,14,15);
            const int nc = wn + f * 16 + mrow;
            v8h b0 = *(const v8h*)&Bs[buf][nc][kb];
            v8h b1 = *(const v8h*)&Bs[buf][nc][kb + 16];
            bf[f] = __builtin_shufflevector(b0, b1, 0,1,2,3,4,5,6,7,8,9,10,11,12,13,14,15);
        }
        #pragma unroll
        for (int i = 0; i < 2; ++i)
            #pragma unroll
            for (int j = 0; j < 2; ++j)
                acc[i][j] = __builtin_amdgcn_wmma_f32_16x16x32_f16(
                    false, af[i], false, bf[j], (short)0, acc[i][j], false, false);

        // ---- stage the register-held next tile into the other buffer, and
        //      prefetch the tile after that (overlaps the WMMAs above) ----
        if (kt + 1 < nk) {
            stage_tiles(As[1 - buf], Bs[1 - buf], tid, opB, ra, rb);
            if (kt + 2 < nk)
                g_load_tiles(A, Bm, M, N, K, lda, ldb, m0, n0, (kt + 2) * 32, opB, tid, ra, rb);
        }
        __syncthreads();
    }

    // ---- epilogue (C/D layout: VGPR r -> M = r + 8*(lane>=16), N = lane&15) ----
    const int rofs = (lane >> 4) << 3;
    #pragma unroll
    for (int i = 0; i < 2; ++i)
        #pragma unroll
        for (int j = 0; j < 2; ++j)
            #pragma unroll
            for (int r = 0; r < 8; ++r) {
                const int row = m0 + wm + i * 16 + rofs + r;
                const int col = n0 + wn + j * 16 + (lane & 15);
                if (row < M && col < N) {
                    float v = acc[i][j][r];
                    if (bias) v += bias[col];
                    if (res)  v += res[(long long)row * ldres + col];
                    if (epi == 1)      v = v > 0.f ? v : (expf(v) - 1.f);
                    else if (epi == 2) v = v > 0.f ? v : 0.f;
                    else if (epi == 3) {
                        v = 1.f / (1.f + expf(-v * oscale));
                        if (mask) v *= (float)mask[(long long)row * ldmask + col];
                    }
                    const long long ci = (long long)bz * sC + (long long)row * ldc + col;
                    if (out16) ((f16*)Cp)[ci] = (f16)v;
                    else       ((float*)Cp)[ci] = v;
                }
            }
}

// ---------------------------------------------------------------------------
// f32 -> f16 convert
// ---------------------------------------------------------------------------
__global__ void k_cvt(const float* __restrict__ s, f16* __restrict__ d, int n) {
    int i = blockIdx.x * 256 + threadIdx.x;
    if (i < n) d[i] = (f16)s[i];
}

// ---------------------------------------------------------------------------
// h1 = elu(x @ W1 + b1), x in {tracks rows (65536), det rows (8192)}, FIN=4
// ---------------------------------------------------------------------------
__global__ __launch_bounds__(256)
void k_feat_h1(const float* __restrict__ tracks, const float* __restrict__ dets,
               const float* __restrict__ W1, const float* __restrict__ b1,
               f16* __restrict__ h1)
{
    const int row = blockIdx.x;          // 0..73727
    const int d   = threadIdx.x;         // 0..255
    const float* x = (row < 65536) ? (tracks + (long long)row * 4)
                                   : (dets + (long long)(row - 65536) * 4);
    float s = b1[d];
    #pragma unroll
    for (int f = 0; f < 4; ++f) s += x[f] * W1[f * 256 + d];
    s = s > 0.f ? s : (expf(s) - 1.f);
    h1[(long long)row * 256 + d] = (f16)s;
}

// ---------------------------------------------------------------------------
// LayerNorm over D=256 (one row per block), f32 in -> f16 out
// ---------------------------------------------------------------------------
__global__ __launch_bounds__(256)
void k_layernorm(const float* __restrict__ in, f16* __restrict__ out)
{
    const int r = blockIdx.x, tid = threadIdx.x;
    __shared__ float s1[256], s2[256];
    float v = in[(long long)r * 256 + tid];
    s1[tid] = v; s2[tid] = v * v;
    __syncthreads();
    for (int st = 128; st > 0; st >>= 1) {
        if (tid < st) { s1[tid] += s1[tid + st]; s2[tid] += s2[tid + st]; }
        __syncthreads();
    }
    const float mean = s1[0] * (1.f / 256.f);
    const float var  = s2[0] * (1.f / 256.f) - mean * mean;
    const float rstd = rsqrtf(var + 1e-5f);
    out[(long long)r * 256 + tid] = (f16)((v - mean) * rstd);
}

// ---------------------------------------------------------------------------
// Per-track attention: R=8 tokens, one (track, head) per block of 64 threads
// qkv rows: (t*8+r)*768 with q/k/v at col c*256 + h*64 + d
// ---------------------------------------------------------------------------
__global__ __launch_bounds__(64)
void k_attention(const f16* __restrict__ qkv, f16* __restrict__ outp)
{
    const int t = blockIdx.x, h = blockIdx.y, tid = threadIdx.x;
    __shared__ float qs[8][64], ks[8][64], vs[8][64], att[8][8];
    const f16* base = qkv + (long long)t * 8 * 768 + h * 64;
    for (int i = tid; i < 512; i += 64) {
        const int r = i >> 6, d = i & 63;
        const f16* rowp = base + (long long)r * 768;
        qs[r][d] = (float)rowp[d];
        ks[r][d] = (float)rowp[256 + d];
        vs[r][d] = (float)rowp[512 + d];
    }
    __syncthreads();
    { // scores 8x8
        const int r = tid >> 3, s = tid & 7;
        float a = 0.f;
        for (int d = 0; d < 64; ++d) a += qs[r][d] * ks[s][d];
        att[r][s] = a * 0.125f;  // 1/sqrt(64)
    }
    __syncthreads();
    if (tid < 8) {
        float mx = att[tid][0];
        for (int s = 1; s < 8; ++s) mx = fmaxf(mx, att[tid][s]);
        float sm = 0.f;
        for (int s = 0; s < 8; ++s) { float e = expf(att[tid][s] - mx); att[tid][s] = e; sm += e; }
        const float inv = 1.f / sm;
        for (int s = 0; s < 8; ++s) att[tid][s] *= inv;
    }
    __syncthreads();
    const int d = tid;
    for (int r = 0; r < 8; ++r) {
        float a = 0.f;
        for (int s = 0; s < 8; ++s) a += att[r][s] * vs[s][d];
        outp[(long long)(t * 8 + r) * 256 + h * 64 + d] = (f16)a;
    }
}

// ---------------------------------------------------------------------------
// predicted -> feat0 (track half): pred = tr_emb[token0] + 0.9*enc0
// ---------------------------------------------------------------------------
__global__ __launch_bounds__(256)
void k_pred(const float* __restrict__ emb, const float* __restrict__ enc0,
            f16* __restrict__ feat0, int trackBase)
{
    const int t = blockIdx.x, d = threadIdx.x;
    const int gt = trackBase + t;
    const int b = gt >> 9, m = gt & 511;
    float v = emb[(long long)gt * 8 * 256 + d] + 0.9f * enc0[(long long)t * 256 + d];
    feat0[((long long)(b * 1024 + m)) * 256 + d] = (f16)v;
}

// det_emb -> feat0 (det half) and featd
__global__ __launch_bounds__(256)
void k_det_feat0(const float* __restrict__ emb, f16* __restrict__ feat0, f16* __restrict__ featd)
{
    const int rn = blockIdx.x;           // b*512 + n
    const int d  = threadIdx.x;
    const int b = rn >> 9, n = rn & 511;
    const f16 v = (f16)emb[(long long)(65536 + rn) * 256 + d];
    feat0[((long long)(b * 1024 + 512 + n)) * 256 + d] = v;
    featd[(long long)rn * 256 + d] = v;
}

// ---------------------------------------------------------------------------
// Adjacency (B,1024,1024) u8: dist^2 < 4 and both nodes valid
// ---------------------------------------------------------------------------
__global__ __launch_bounds__(256)
void k_adj(const float* __restrict__ tracks, const float* __restrict__ dets,
           const int* __restrict__ tmarks, const int* __restrict__ dmarks,
           unsigned char* __restrict__ adj)
{
    const int i = blockIdx.x, b = blockIdx.y, tid = threadIdx.x;
    float xi, yi; bool vi;
    if (i < 512) {
        const long long o = ((long long)(b * 512 + i) * 8) * 4;
        xi = tracks[o]; yi = tracks[o + 1]; vi = i < tmarks[b];
    } else {
        const long long o = (long long)(b * 512 + (i - 512)) * 4;
        xi = dets[o]; yi = dets[o + 1]; vi = (i - 512) < dmarks[b];
    }
    unsigned char* row = adj + (long long)b * 1048576 + (long long)i * 1024;
    for (int j = tid; j < 1024; j += 256) {
        float xj, yj; bool vj;
        if (j < 512) {
            const long long o = ((long long)(b * 512 + j) * 8) * 4;
            xj = tracks[o]; yj = tracks[o + 1]; vj = j < tmarks[b];
        } else {
            const long long o = (long long)(b * 512 + (j - 512)) * 4;
            xj = dets[o]; yj = dets[o + 1]; vj = (j - 512) < dmarks[b];
        }
        const float dx = xi - xj, dy = yi - yj;
        row[j] = (unsigned char)((vi && vj && (dx * dx + dy * dy < 4.f)) ? 1 : 0);
    }
}

// ---------------------------------------------------------------------------
// src/dst per GAT head: src[row][h] = sum_d h[row][h*64+d]*a0[h*64+d]
// ---------------------------------------------------------------------------
__global__ __launch_bounds__(256)
void k_srcdst(const f16* __restrict__ h, const float* __restrict__ a,
              float* __restrict__ src, float* __restrict__ dst)
{
    const int row = blockIdx.x, tid = threadIdx.x;
    __shared__ float s0[256], s1[256];
    const float hv = (float)h[(long long)row * 256 + tid];
    s0[tid] = hv * a[tid];
    s1[tid] = hv * a[256 + tid];
    __syncthreads();
    for (int st = 32; st > 0; st >>= 1) {
        if ((tid & 63) < st) { s0[tid] += s0[tid + st]; s1[tid] += s1[tid + st]; }
        __syncthreads();
    }
    if ((tid & 63) == 0) {
        const int hh = tid >> 6;
        src[(long long)row * 4 + hh] = s0[tid];
        dst[(long long)row * 4 + hh] = s1[tid];
    }
}

// ---------------------------------------------------------------------------
// Fused GAT edge-softmax + aggregation + elu + residual.
// One block per (node i, batch b). alpha[h][j] lives in LDS.
// ---------------------------------------------------------------------------
__global__ __launch_bounds__(256)
void k_gat_agg(const f16* __restrict__ h, const float* __restrict__ src,
               const float* __restrict__ dst,
               const unsigned char* __restrict__ adjBase, int adjLd, long long adjStride,
               const f16* __restrict__ f0, f16* __restrict__ xout,
               int NN, float wG, float wO)
{
    const int i = blockIdx.x, b = blockIdx.y, tid = threadIdx.x;
    __shared__ float alpha[4][1024];
    __shared__ float red[256][4];

    const f16*   hb   = h   + (long long)b * NN * 256;
    const float* srcb = src + (long long)b * NN * 4;
    const float* dstb = dst + (long long)b * NN * 4;
    const unsigned char* arow = adjBase + (long long)b * adjStride + (long long)i * adjLd;

    float sv[4];
    #pragma unroll
    for (int hh = 0; hh < 4; ++hh) sv[hh] = srcb[(long long)i * 4 + hh];

    float lmax[4] = {-1e30f, -1e30f, -1e30f, -1e30f};
    for (int j = tid; j < NN; j += 256) {
        const bool on = arow[j] != 0;
        #pragma unroll
        for (int hh = 0; hh < 4; ++hh) {
            float e;
            if (on) { e = sv[hh] + dstb[(long long)j * 4 + hh]; e = e < 0.f ? 0.2f * e : e; }
            else e = -1e9f;
            alpha[hh][j] = e;
            lmax[hh] = fmaxf(lmax[hh], e);
        }
    }
    #pragma unroll
    for (int hh = 0; hh < 4; ++hh) red[tid][hh] = lmax[hh];
    __syncthreads();
    for (int st = 128; st > 0; st >>= 1) {
        if (tid < st)
            #pragma unroll
            for (int hh = 0; hh < 4; ++hh) red[tid][hh] = fmaxf(red[tid][hh], red[tid + st][hh]);
        __syncthreads();
    }
    float mx[4];
    #pragma unroll
    for (int hh = 0; hh < 4; ++hh) mx[hh] = red[0][hh];
    __syncthreads();

    float lsum[4] = {0.f, 0.f, 0.f, 0.f};
    for (int j = tid; j < NN; j += 256) {
        #pragma unroll
        for (int hh = 0; hh < 4; ++hh) {
            const float e = expf(alpha[hh][j] - mx[hh]);
            alpha[hh][j] = e;
            lsum[hh] += e;
        }
    }
    #pragma unroll
    for (int hh = 0; hh < 4; ++hh) red[tid][hh] = lsum[hh];
    __syncthreads();
    for (int st = 128; st > 0; st >>= 1) {
        if (tid < st)
            #pragma unroll
            for (int hh = 0; hh < 4; ++hh) red[tid][hh] += red[tid + st][hh];
        __syncthreads();
    }
    float inv[4];
    #pragma unroll
    for (int hh = 0; hh < 4; ++hh) inv[hh] = 1.f / red[0][hh];
    for (int j = tid; j < NN; j += 256)
        #pragma unroll
        for (int hh = 0; hh < 4; ++hh) alpha[hh][j] *= inv[hh];
    __syncthreads();

    // aggregation: thread = output dim d, head = d>>6
    const int d = tid, hh = d >> 6;
    float acc = 0.f;
    for (int j = 0; j < NN; ++j)
        acc += alpha[hh][j] * (float)hb[(long long)j * 256 + d];
    const float eluv = acc > 0.f ? acc : (expf(acc) - 1.f);
    const float outv = wG * eluv + wO * (float)f0[(long long)(b * NN + i) * 256 + d];
    xout[(long long)(b * NN + i) * 256 + d] = (f16)outv;
}

// asso = scores[:, :512, 512:]
__global__ __launch_bounds__(256)
void k_copy_asso(const float* __restrict__ scores, float* __restrict__ asso)
{
    const int m = blockIdx.x, b = blockIdx.y, tid = threadIdx.x;
    for (int n = tid; n < 512; n += 256)
        asso[(long long)b * 262144 + (long long)m * 512 + n] =
            scores[(long long)b * 1048576 + (long long)m * 1024 + 512 + n];
}

__global__ __launch_bounds__(512)
void k_zero_diag(float* __restrict__ ds)
{
    ds[(long long)blockIdx.x * 262144 + (long long)threadIdx.x * 512 + threadIdx.x] = 0.f;
}

// ---------------------------------------------------------------------------
// Orchestration
// ---------------------------------------------------------------------------
extern "C" void kernel_launch(void* const* d_in, const int* in_sizes, int n_in,
                              void* d_out, int out_size, void* d_ws, size_t ws_size,
                              hipStream_t stream)
{
    (void)in_sizes; (void)n_in; (void)out_size; (void)ws_size;
    const float* tracks = (const float*)d_in[0];
    const float* dets   = (const float*)d_in[1];
    const int*   tmarks = (const int*)d_in[2];
    const int*   dmarks = (const int*)d_in[3];
    const float* W1   = (const float*)d_in[4];
    const float* b1   = (const float*)d_in[5];
    const float* W2   = (const float*)d_in[6];
    const float* b2   = (const float*)d_in[7];
    const float* Wqkv = (const float*)d_in[8];
    const float* Wo   = (const float*)d_in[9];
    const float* ff1  = (const float*)d_in[10];
    const float* ff2  = (const float*)d_in[11];
    const float* gatW = (const float*)d_in[12];
    const float* gatA = (const float*)d_in[13];
    const float* dgatW= (const float*)d_in[14];
    const float* dgatA= (const float*)d_in[15];
    const float* clsWq= (const float*)d_in[16];
    const float* clsWk= (const float*)d_in[17];

    float* out_scores  = (float*)d_out;                          // 16*1024*1024
    float* out_dscores = out_scores + 16ll * 1024 * 1024;        // 16*512*512
    float* out_asso    = out_dscores + 16ll * 512 * 512;         // 16*512*512

    char* base = (char*)d_ws; size_t off = 0;
    auto alloc = [&](size_t bytes) -> void* {
        void* p = base + off;
        off += (bytes + 255) & ~(size_t)255;
        return p;
    };

    // f16 weight copies
    f16* W2h    = (f16*)alloc(65536ll * 2);
    f16* Wqkvh  = (f16*)alloc(196608ll * 2);
    f16* Woh    = (f16*)alloc(65536ll * 2);
    f16* ff1h   = (f16*)alloc(262144ll * 2);
    f16* ff2h   = (f16*)alloc(262144ll * 2);
    f16* gatWh  = (f16*)alloc(131072ll * 2);
    f16* dgatWh = (f16*)alloc(131072ll * 2);
    f16* clsWqh = (f16*)alloc(131072ll * 2);
    f16* clsWkh = (f16*)alloc(131072ll * 2);
    // activations
    f16*   h1    = (f16*)alloc(73728ll * 256 * 2);
    float* emb   = (float*)alloc(73728ll * 256 * 4);
    f16*   feat0 = (f16*)alloc(16ll * 1024 * 256 * 2);
    f16*   featd = (f16*)alloc(16ll * 512 * 256 * 2);
    unsigned char* adj = (unsigned char*)alloc(16ll * 1024 * 1024);
    f16*   xbuf  = (f16*)alloc(16ll * 1024 * 256 * 2);
    f16*   hgat  = (f16*)alloc(16ll * 1024 * 256 * 2);
    float* srcb  = (float*)alloc(16ll * 1024 * 4 * 4);
    float* dstb  = (float*)alloc(16ll * 1024 * 4 * 4);
    f16*   qbuf  = (f16*)alloc(16ll * 1024 * 256 * 2);
    f16*   kbuf  = (f16*)alloc(16ll * 1024 * 256 * 2);
    f16*   xdbuf = (f16*)alloc(16ll * 512 * 256 * 2);
    // encoder chunk scratch (2048 tracks = 16384 tokens per chunk)
    f16*   ln1h  = (f16*)alloc(16384ll * 256 * 2);
    f16*   qkvh  = (f16*)alloc(16384ll * 768 * 2);
    f16*   attnh = (f16*)alloc(16384ll * 256 * 2);
    float* x2    = (float*)alloc(16384ll * 256 * 4);
    f16*   ln2h  = (f16*)alloc(16384ll * 256 * 2);
    f16*   ffh   = (f16*)alloc(16384ll * 1024 * 2);
    float* enc0  = (float*)alloc(2048ll * 256 * 4);

    auto gemm = [&](const f16* A, const f16* B, const float* bias_, const float* res_,
                    void* C, const unsigned char* mask_,
                    int M, int N, int K, int lda, int ldb, int ldc, int ldres, int ldmask,
                    int opB, int epi, int out16, float oscale, int batch,
                    long long sA, long long sB, long long sC, long long sRes, long long sMask) {
        dim3 g((N + 63) / 64, (M + 63) / 64, batch), blk(128);
        k_wmma_gemm<<<g, blk, 0, stream>>>(A, B, bias_, res_, C, mask_, M, N, K,
                                           lda, ldb, ldc, ldres, ldmask,
                                           opB, epi, out16, oscale, sA, sB, sC, sRes, sMask);
    };
    auto cvt = [&](const float* s, f16* d, int n) {
        k_cvt<<<(n + 255) / 256, 256, 0, stream>>>(s, d, n);
    };

    cvt(W2, W2h, 65536);       cvt(Wqkv, Wqkvh, 196608);  cvt(Wo, Woh, 65536);
    cvt(ff1, ff1h, 262144);    cvt(ff2, ff2h, 262144);
    cvt(gatW, gatWh, 131072);  cvt(dgatW, dgatWh, 131072);
    cvt(clsWq, clsWqh, 131072); cvt(clsWk, clsWkh, 131072);

    // feature MLP: h1 then emb = h1 @ W2 + b2
    k_feat_h1<<<dim3(73728), 256, 0, stream>>>(tracks, dets, W1, b1, h1);
    gemm(h1, W2h, b2, nullptr, emb, nullptr, 73728, 256, 256, 256, 256, 256, 0, 0,
         0, 0, 0, 1.f, 1, 0, 0, 0, 0, 0);

    k_adj<<<dim3(1024, 16), 256, 0, stream>>>(tracks, dets, tmarks, dmarks, adj);

    // encoder, chunked over 4 x 2048 tracks
    for (int c = 0; c < 4; ++c) {
        const float* embc = emb + (long long)c * 16384 * 256;
        k_layernorm<<<16384, 256, 0, stream>>>(embc, ln1h);
        gemm(ln1h, Wqkvh, nullptr, nullptr, qkvh, nullptr, 16384, 768, 256,
             256, 768, 768, 0, 0, 0, 0, 1, 1.f, 1, 0, 0, 0, 0, 0);
        k_attention<<<dim3(2048, 4), 64, 0, stream>>>(qkvh, attnh);
        gemm(attnh, Woh, nullptr, embc, x2, nullptr, 16384, 256, 256,
             256, 256, 256, 256, 0, 0, 0, 0, 1.f, 1, 0, 0, 0, 0, 0);
        k_layernorm<<<16384, 256, 0, stream>>>(x2, ln2h);
        gemm(ln2h, ff1h, nullptr, nullptr, ffh, nullptr, 16384, 1024, 256,
             256, 1024, 1024, 0, 0, 0, 2, 1, 1.f, 1, 0, 0, 0, 0, 0);
        // only token-0 rows needed downstream: stride A/res by 8 rows
        gemm(ffh, ff2h, nullptr, x2, enc0, nullptr, 2048, 256, 1024,
             8192, 256, 256, 2048, 0, 0, 0, 0, 1.f, 1, 0, 0, 0, 0, 0);
        k_pred<<<2048, 256, 0, stream>>>(emb, enc0, feat0, c * 2048);
    }
    k_det_feat0<<<8192, 256, 0, stream>>>(emb, feat0, featd);

    // main GAT (2 layers over 16 x 1024 nodes)
    for (int l = 0; l < 2; ++l) {
        const f16* xin = (l == 0) ? feat0 : xbuf;
        gemm(xin, gatWh + l * 65536, nullptr, nullptr, hgat, nullptr, 16384, 256, 256,
             256, 256, 256, 0, 0, 0, 0, 1, 1.f, 1, 0, 0, 0, 0, 0);
        k_srcdst<<<16384, 256, 0, stream>>>(hgat, gatA + l * 512, srcb, dstb);
        k_gat_agg<<<dim3(1024, 16), 256, 0, stream>>>(hgat, srcb, dstb, adj, 1024, 1048576ll,
                                                      feat0, xbuf, 1024, 0.5f, 0.5f);
    }
    // scores = sigmoid((x Wq)(x Wk)^T / 16) * adj  -> written directly into d_out
    gemm(xbuf, clsWqh, nullptr, nullptr, qbuf, nullptr, 16384, 256, 256,
         256, 256, 256, 0, 0, 0, 0, 1, 1.f, 1, 0, 0, 0, 0, 0);
    gemm(xbuf, clsWkh, nullptr, nullptr, kbuf, nullptr, 16384, 256, 256,
         256, 256, 256, 0, 0, 0, 0, 1, 1.f, 1, 0, 0, 0, 0, 0);
    gemm(qbuf, kbuf, nullptr, nullptr, out_scores, adj, 1024, 1024, 256,
         256, 256, 1024, 0, 1024, 1, 3, 0, 0.0625f, 16,
         1024ll * 256, 1024ll * 256, 1024ll * 1024, 0, 1048576ll);
    k_copy_asso<<<dim3(512, 16), 256, 0, stream>>>(out_scores, out_asso);

    // detection GAT (2 layers over 16 x 512 nodes); adj_d == adj[:,512:,512:]
    const unsigned char* adjd = adj + 512 * 1024 + 512;
    for (int l = 0; l < 2; ++l) {
        const f16* xin = (l == 0) ? featd : xdbuf;
        gemm(xin, dgatWh + l * 65536, nullptr, nullptr, hgat, nullptr, 8192, 256, 256,
             256, 256, 256, 0, 0, 0, 0, 1, 1.f, 1, 0, 0, 0, 0, 0);
        k_srcdst<<<8192, 256, 0, stream>>>(hgat, dgatA + l * 512, srcb, dstb);
        k_gat_agg<<<dim3(512, 16), 256, 0, stream>>>(hgat, srcb, dstb, adjd, 1024, 1048576ll,
                                                     featd, xdbuf, 512, 0.5f, 0.5f);
    }
    gemm(xdbuf, clsWqh + 65536, nullptr, nullptr, qbuf, nullptr, 8192, 256, 256,
         256, 256, 256, 0, 0, 0, 0, 1, 1.f, 1, 0, 0, 0, 0, 0);
    gemm(xdbuf, clsWkh + 65536, nullptr, nullptr, kbuf, nullptr, 8192, 256, 256,
         256, 256, 256, 0, 0, 0, 0, 1, 1.f, 1, 0, 0, 0, 0, 0);
    gemm(qbuf, kbuf, nullptr, nullptr, out_dscores, adjd, 512, 512, 256,
         256, 256, 512, 0, 1024, 1, 3, 0, 0.0625f, 16,
         512ll * 256, 512ll * 256, 512ll * 512, 0, 1048576ll);
    k_zero_diag<<<16, 512, 0, stream>>>(out_dscores);
}
